// Attention_6614249636041
// MI455X (gfx1250) — matrix operand order repaired and checked by reading, not compiled, not run on hardware
//
#include <hip/hip_runtime.h>
#include <hip/hip_bf16.h>
#include <stdint.h>

// Problem constants (from the reference)
#define HIDDEN 2048
#define NHEADS 32
#define NKV 8
#define HEAD 64
#define BATCH 2
#define SEQ 2048
#define M_TOTAL (BATCH * SEQ)  // 4096 rows of x flattened
#define KVDIM (NKV * HEAD)     // 512
#define LDSP 72                // padded LDS row stride (144 B, 16-B aligned)

typedef __bf16 bf16_t;
typedef __attribute__((ext_vector_type(16))) __bf16 v16bf;
typedef __attribute__((ext_vector_type(8)))  __bf16 v8bf;
typedef __attribute__((ext_vector_type(8)))  float  v8f;

// ---------------------------------------------------------------------------
// WMMA helper: D = A(16x32 bf16) * B(32x16 bf16) + C(16x16 f32)
// ---------------------------------------------------------------------------
__device__ __forceinline__ v8f wmma_bf16(v16bf a, v16bf b, v8f c) {
  return __builtin_amdgcn_wmma_f32_16x16x32_bf16(
      /*neg_a=*/false, a, /*neg_b=*/false, b,
      /*c_mod=*/(short)0, c, /*reuse_a=*/false, /*reuse_b=*/false);
}

// A-operand fragment (16x32, MxK). ISA layout: lane l holds row m=l%16,
// K-chunks [k0,k0+8) and [k0+16,k0+24), k0=(l/16)*8. Two 16-byte loads.
__device__ __forceinline__ v16bf load_a_frag(const bf16_t* base, int ld, int lane) {
  const int m  = lane & 15;
  const int k0 = (lane >> 4) << 3;
  const v8bf lo = *reinterpret_cast<const v8bf*>(base + (size_t)m * ld + k0);
  const v8bf hi = *reinterpret_cast<const v8bf*>(base + (size_t)m * ld + k0 + 16);
  v16bf r;
#pragma unroll
  for (int i = 0; i < 8; ++i) { r[i] = lo[i]; r[i + 8] = hi[i]; }
  return r;
}

// B-operand fragment (32x16, KxN) supplied as Wt[n][k] row-major (K contiguous):
// lane l holds column n=l%16, K-chunks [k0,k0+8) and [k0+16,k0+24), k0=(l/16)*8.
// Two 16-byte loads (legal for 16-B aligned padded LDS rows too).
__device__ __forceinline__ v16bf load_b_frag(const bf16_t* base, int ld, int lane) {
  const int n  = lane & 15;
  const int k0 = (lane >> 4) << 3;
  const v8bf lo = *reinterpret_cast<const v8bf*>(base + (size_t)n * ld + k0);
  const v8bf hi = *reinterpret_cast<const v8bf*>(base + (size_t)n * ld + k0 + 16);
  v16bf r;
#pragma unroll
  for (int i = 0; i < 8; ++i) { r[i] = lo[i]; r[i + 8] = hi[i]; }
  return r;
}

// ---------------------------------------------------------------------------
// f32 -> bf16 conversion (vectorized x4)
// ---------------------------------------------------------------------------
__global__ __launch_bounds__(256)
void cvt_f32_bf16(const float* __restrict__ s, bf16_t* __restrict__ d, long n) {
  long i = ((long)blockIdx.x * blockDim.x + threadIdx.x) * 4;
  const long stride = (long)gridDim.x * blockDim.x * 4;
  for (; i < n; i += stride) {
    const float4 f = *reinterpret_cast<const float4*>(s + i);
    union { bf16_t h[4]; uint64_t u; } pk;
    pk.h[0] = (bf16_t)f.x; pk.h[1] = (bf16_t)f.y;
    pk.h[2] = (bf16_t)f.z; pk.h[3] = (bf16_t)f.w;
    *reinterpret_cast<uint64_t*>(d + i) = pk.u;
  }
}

// ---------------------------------------------------------------------------
// GEMM: C[M,N] = A[M,K] * W[N,K]^T  (torch Linear), bf16 inputs, f32 accum.
// Block = 256 threads (8 waves), 128x128 C tile; wave = 32(M) x 64(N).
// Operands fetched directly from global: the whole bf16 working set (~40 MB)
// is resident in the 192 MB L2, and WGP-scope prefetch pulls upcoming K tiles
// into the near caches.
// ---------------------------------------------------------------------------
__global__ __launch_bounds__(256)
void gemm_bf16_nt(const bf16_t* __restrict__ A, const bf16_t* __restrict__ W,
                  const float* __restrict__ bias,
                  float* __restrict__ Cf, bf16_t* __restrict__ Cb,
                  int M, int N, int K) {
  const int lane = threadIdx.x & 31;
  const int wave = threadIdx.x >> 5;                 // 0..7
  const int row0 = blockIdx.y * 128 + (wave >> 1) * 32;
  const int col0 = blockIdx.x * 128 + (wave & 1) * 64;

  v8f acc[2][4];
#pragma unroll
  for (int mi = 0; mi < 2; ++mi)
#pragma unroll
    for (int t = 0; t < 4; ++t)
#pragma unroll
      for (int e = 0; e < 8; ++e) acc[mi][t][e] = 0.0f;

  const bf16_t* a0p = A + (size_t)row0 * K;
  const bf16_t* a1p = A + ((size_t)row0 + 16) * K;

  for (int k0 = 0; k0 < K; k0 += 32) {
    if (k0 + 256 < K) {  // WGP-scope prefetch (locality 3): fill near caches
      __builtin_prefetch(a0p + k0 + 256, 0, 3);
      __builtin_prefetch(W + (size_t)col0 * K + k0 + 256, 0, 3);
    }
    const v16bf a0 = load_a_frag(a0p + k0, K, lane);
    const v16bf a1 = load_a_frag(a1p + k0, K, lane);
#pragma unroll
    for (int t = 0; t < 4; ++t) {
      const v16bf bt = load_b_frag(W + (size_t)(col0 + t * 16) * K + k0, K, lane);
      acc[0][t] = wmma_bf16(a0, bt, acc[0][t]);
      acc[1][t] = wmma_bf16(a1, bt, acc[1][t]);
    }
  }

  // C/D f32 layout: element [m][n] lives in vgpr m%8, lane n + 16*(m/8)
  const int moff = (lane >> 4) << 3;
  const int n0   = lane & 15;
#pragma unroll
  for (int mi = 0; mi < 2; ++mi)
#pragma unroll
    for (int t = 0; t < 4; ++t)
#pragma unroll
      for (int r = 0; r < 8; ++r) {
        const size_t m = (size_t)row0 + mi * 16 + r + moff;
        const int    n = col0 + t * 16 + n0;
        float v = acc[mi][t][r];
        if (bias) v += bias[n];
        if (Cf) Cf[m * (size_t)N + n] = v;
        else    Cb[m * (size_t)N + n] = (bf16_t)v;
      }
}

// ---------------------------------------------------------------------------
// Fused flash-attention (non-causal), bf16 WMMA, f32 online softmax.
// Grid: (SEQ/64, BATCH*NHEADS). Block = 128 threads (4 waves); each wave owns
// 16 query rows. Per 64-key chunk:
//  - the K chunk is staged ONCE per block into LDS with the CDNA5 async path
//    (global_load_async_to_lds_b128, ASYNCcnt), removing the 4x per-wave
//    redundant global fetch; the copy overlaps the V transpose.
//  - S = scale*Q*K^T (WMMA from LDS), online softmax with wave32 shuffles
//    (each score row lives in one 16-lane half per the ISA D-layout),
//  - P relaid to the A-operand layout via LDS (wave-local, DS ops are
//    in-order per wave), V^T staged in LDS for the B operand, O += P*V.
// ---------------------------------------------------------------------------
__global__ __launch_bounds__(128)
void attn_fused(const bf16_t* __restrict__ Q, const bf16_t* __restrict__ Kb,
                const bf16_t* __restrict__ Vb, bf16_t* __restrict__ O) {
  const float scale = 0.125f;  // HEAD^-0.5 = 1/8
  const int lane = threadIdx.x & 31;
  const int wave = threadIdx.x >> 5;  // 0..3
  const int qt = blockIdx.x;          // 0..SEQ/64-1
  const int bh = blockIdx.y;          // 0..BATCH*NHEADS-1
  const int b  = bh >> 5;
  const int h  = bh & 31;
  const int hkv = h >> 2;             // GQA: 4 heads share one kv head

  __shared__ bf16_t Kls[64][LDSP];     // K chunk, row-major [key][d]
  __shared__ bf16_t Vt[64][LDSP];      // V^T chunk [d][key]
  __shared__ bf16_t Pls[4][16][LDSP];  // per-wave P tile [m][key]

  const size_t qrow0 = (size_t)b * SEQ + (size_t)qt * 64 + wave * 16;
  const bf16_t* qp = Q  + qrow0 * HIDDEN + h * HEAD;
  const bf16_t* kp = Kb + (size_t)b * SEQ * KVDIM + hkv * HEAD;
  const bf16_t* vp = Vb + (size_t)b * SEQ * KVDIM + hkv * HEAD;

  // Q fragments for both 32-wide K(=D) steps; reused across the whole key loop
  v16bf qf[2];
  qf[0] = load_a_frag(qp,      HIDDEN, lane);
  qf[1] = load_a_frag(qp + 32, HIDDEN, lane);

  float mrow[8], lrow[8];
  v8f o[4];
#pragma unroll
  for (int r = 0; r < 8; ++r) { mrow[r] = -1e30f; lrow[r] = 0.0f; }
#pragma unroll
  for (int t = 0; t < 4; ++t)
#pragma unroll
    for (int e = 0; e < 8; ++e) o[t][e] = 0.0f;

  for (int kc = 0; kc < SEQ; kc += 64) {
    __syncthreads();  // previous iteration's LDS consumers are done

    // ---- async-stage the 64x64 K chunk into LDS (16 B per lane per issue;
    //      4 issues x 128 threads = 8 KB). Hardware adds the wave LDS base.
#pragma unroll
    for (int i = 0; i < 4; ++i) {
      const int idx = threadIdx.x + i * 128;   // 0..511
      const int row = idx >> 3;                // key row 0..63
      const int c16 = (idx & 7) * 8;           // 8 bf16 = 16 B column chunk
      const uint64_t gaddr =
          (uint64_t)(uintptr_t)(kp + (size_t)(kc + row) * KVDIM + c16);
      const uint32_t laddr = (uint32_t)(uintptr_t)(&Kls[row][c16]);
      asm volatile("global_load_async_to_lds_b128 %0, %1, off"
                   :: "v"(laddr), "v"(gaddr) : "memory");
    }

    // ---- cooperative transpose of the 64x64 V chunk into LDS (overlaps the
    //      async K copy; coalesced 128-bit global reads, scatter b16 stores)
    {
      const int row = threadIdx.x >> 1;
      const int c0  = (threadIdx.x & 1) * 32;
      const bf16_t* src = vp + (size_t)(kc + row) * KVDIM + c0;
#pragma unroll
      for (int c = 0; c < 32; ++c) Vt[c0 + c][row] = src[c];
    }

    asm volatile("s_wait_asynccnt 0x0" ::: "memory");
    __syncthreads();  // Kls + Vt ready for all waves

    // ---- scores: S[16q x 64key] = Q * K^T (B operand = K rows from LDS) ----
    v8f s[4];
#pragma unroll
    for (int t = 0; t < 4; ++t) {
#pragma unroll
      for (int e = 0; e < 8; ++e) s[t][e] = 0.0f;
#pragma unroll
      for (int kk = 0; kk < 2; ++kk) {
        const v16bf kf = load_b_frag(&Kls[t * 16][0] + kk * 32, LDSP, lane);
        s[t] = wmma_bf16(qf[kk], kf, s[t]);
      }
    }

    // ---- online softmax: row m = r + 8*(lane/16) lives in one 16-lane half
    float ss[4][8];
#pragma unroll
    for (int t = 0; t < 4; ++t)
#pragma unroll
      for (int r = 0; r < 8; ++r) ss[t][r] = s[t][r] * scale;

    float rmax[8];
#pragma unroll
    for (int r = 0; r < 8; ++r) {
      float v = ss[0][r];
#pragma unroll
      for (int t = 1; t < 4; ++t) v = fmaxf(v, ss[t][r]);
      rmax[r] = v;
    }
#pragma unroll
    for (int off = 8; off >= 1; off >>= 1)
#pragma unroll
      for (int r = 0; r < 8; ++r)
        rmax[r] = fmaxf(rmax[r], __shfl_xor(rmax[r], off, 32));

    float alpha[8];
#pragma unroll
    for (int r = 0; r < 8; ++r) {
      const float mn = fmaxf(mrow[r], rmax[r]);
      alpha[r] = __expf(mrow[r] - mn);
      mrow[r] = mn;
    }

    // P = exp(S - m), stash bf16 tile in LDS (A-layout source), track row sums
    float rsum[8];
#pragma unroll
    for (int r = 0; r < 8; ++r) rsum[r] = 0.0f;
    const int mbase = (lane >> 4) << 3;
    const int cbase = lane & 15;
#pragma unroll
    for (int t = 0; t < 4; ++t)
#pragma unroll
      for (int r = 0; r < 8; ++r) {
        const float p = __expf(ss[t][r] - mrow[r]);
        rsum[r] += p;
        Pls[wave][r + mbase][t * 16 + cbase] = (bf16_t)p;
      }
#pragma unroll
    for (int off = 8; off >= 1; off >>= 1)
#pragma unroll
      for (int r = 0; r < 8; ++r) rsum[r] += __shfl_xor(rsum[r], off, 32);
#pragma unroll
    for (int r = 0; r < 8; ++r) lrow[r] = lrow[r] * alpha[r] + rsum[r];
#pragma unroll
    for (int td = 0; td < 4; ++td)
#pragma unroll
      for (int r = 0; r < 8; ++r) o[td][r] *= alpha[r];

    // ---- O += P * V  (A = P, B = V^T rows; both from LDS; P is wave-local
    //      and DS ops are in-order per wave, so no extra barrier needed) ----
#pragma unroll
    for (int kk = 0; kk < 2; ++kk) {
      const v16bf pf = load_a_frag(&Pls[wave][0][0] + kk * 32, LDSP, lane);
#pragma unroll
      for (int td = 0; td < 4; ++td) {
        const v16bf vf = load_b_frag(&Vt[td * 16][0] + kk * 32, LDSP, lane);
        o[td] = wmma_bf16(pf, vf, o[td]);
      }
    }
  }

  // normalize and write attention output (bf16, feeds the Wo GEMM)
#pragma unroll
  for (int r = 0; r < 8; ++r) {
    const float inv = 1.0f / lrow[r];
    const size_t row = qrow0 + r + ((lane >> 4) << 3);
#pragma unroll
    for (int td = 0; td < 4; ++td) {
      const int col = h * HEAD + td * 16 + (lane & 15);
      O[row * HIDDEN + col] = (bf16_t)(o[td][r] * inv);
    }
  }
}

// ---------------------------------------------------------------------------
// Launch: cvt(x, Wq, Wk, Wv, Wo) -> q/k/v GEMMs -> fused attention -> out GEMM
// Workspace (bf16 elements): ~76 MB total.
// ---------------------------------------------------------------------------
extern "C" void kernel_launch(void* const* d_in, const int* in_sizes, int n_in,
                              void* d_out, int out_size, void* d_ws, size_t ws_size,
                              hipStream_t stream) {
  (void)in_sizes; (void)n_in; (void)out_size; (void)ws_size;
  const float* x  = (const float*)d_in[0];
  const float* Wq = (const float*)d_in[1];
  const float* Wk = (const float*)d_in[2];
  const float* Wv = (const float*)d_in[3];
  const float* Wo = (const float*)d_in[4];
  const float* bo = (const float*)d_in[5];

  const long nX   = (long)M_TOTAL * HIDDEN;  // 8388608
  const long nWq  = (long)HIDDEN * HIDDEN;   // 4194304
  const long nWkv = (long)KVDIM * HIDDEN;    // 1048576
  const long nKV  = (long)M_TOTAL * KVDIM;   // 2097152

  bf16_t* w   = (bf16_t*)d_ws;
  bf16_t* xb  = w; w += nX;
  bf16_t* wqb = w; w += nWq;
  bf16_t* wkb = w; w += nWkv;
  bf16_t* wvb = w; w += nWkv;
  bf16_t* wob = w; w += nWq;
  bf16_t* qb  = w; w += nX;
  bf16_t* kb  = w; w += nKV;
  bf16_t* vb  = w; w += nKV;
  bf16_t* ab  = w; w += nX;

  auto cvt = [&](const float* s, bf16_t* d, long n) {
    const int blocks = (int)((n / 4 + 255) / 256);
    cvt_f32_bf16<<<blocks, 256, 0, stream>>>(s, d, n);
  };
  cvt(x,  xb,  nX);
  cvt(Wq, wqb, nWq);
  cvt(Wk, wkb, nWkv);
  cvt(Wv, wvb, nWkv);
  cvt(Wo, wob, nWq);

  // q = x@Wq^T, k = x@Wk^T, v = x@Wv^T  (bf16 outputs into workspace)
  gemm_bf16_nt<<<dim3(HIDDEN / 128, M_TOTAL / 128), 256, 0, stream>>>(
      xb, wqb, nullptr, nullptr, qb, M_TOTAL, HIDDEN, HIDDEN);
  gemm_bf16_nt<<<dim3(KVDIM / 128, M_TOTAL / 128), 256, 0, stream>>>(
      xb, wkb, nullptr, nullptr, kb, M_TOTAL, KVDIM, HIDDEN);
  gemm_bf16_nt<<<dim3(KVDIM / 128, M_TOTAL / 128), 256, 0, stream>>>(
      xb, wvb, nullptr, nullptr, vb, M_TOTAL, KVDIM, HIDDEN);

  // fused softmax(QK^T/sqrt(d))V with GQA head sharing
  attn_fused<<<dim3(SEQ / 64, BATCH * NHEADS), 128, 0, stream>>>(qb, kb, vb, ab);

  // out = attn@Wo^T + bo  (f32 output)
  gemm_bf16_nt<<<dim3(HIDDEN / 128, M_TOTAL / 128), 256, 0, stream>>>(
      ab, wob, bo, (float*)d_out, nullptr, M_TOTAL, HIDDEN, HIDDEN);
}
